// ROIClassifierHead_69836168233060
// MI455X (gfx1250) — matrix operand order repaired and checked by reading, not compiled, not run on hardware
//
#include <hip/hip_runtime.h>
#include <stdint.h>

typedef __attribute__((ext_vector_type(16))) __bf16 v16bf;
typedef __attribute__((ext_vector_type(8)))  float  v8f;

#define BM 128
#define BN 256
#define BK 32
#define LDSP 40   // padded LDS row stride in ushorts (80B, 16B aligned)

__device__ __forceinline__ unsigned short f2bf(float f) {
    unsigned int u = __float_as_uint(f);
    u = (u + 0x7FFFu + ((u >> 16) & 1u)) >> 16;   // round to nearest even
    return (unsigned short)u;
}

// Async global->LDS 16B copy, SADDR form: base in SGPRs, 32-bit byte offset in
// VGPR. INST_OFFSET is added to BOTH the LDS and the global address (ISA 8.4.4),
// so consecutive 16B chunks need no extra address math.
template<int OFF>
__device__ __forceinline__ void async_copy_b128(unsigned lds_byte_addr,
                                                unsigned gbyte_off,
                                                const void* gbase) {
    asm volatile("global_load_async_to_lds_b128 %0, %1, %2 offset:%c3"
                 :
                 : "v"(lds_byte_addr), "v"(gbyte_off), "s"(gbase), "i"(OFF)
                 : "memory");
}
__device__ __forceinline__ void wait_async() {
    asm volatile("s_wait_asynccnt 0" ::: "memory");
}
// low 32 bits of a generic pointer to __shared__ == group-segment byte offset
__device__ __forceinline__ unsigned lds_off(const void* p) {
    return (unsigned)(uintptr_t)p;
}

// ---------------- f32 -> bf16 conversion (vectorized x4) ----------------
__global__ void cvt_f32_bf16_kernel(const float* __restrict__ src,
                                    unsigned short* __restrict__ dst,
                                    long long n4) {
    long long i = (long long)blockIdx.x * blockDim.x + threadIdx.x;
    long long stride = (long long)gridDim.x * blockDim.x;
    for (; i < n4; i += stride) {
        float4 f = reinterpret_cast<const float4*>(src)[i];
        ushort4 o;
        o.x = f2bf(f.x); o.y = f2bf(f.y); o.z = f2bf(f.z); o.w = f2bf(f.w);
        reinterpret_cast<ushort4*>(dst)[i] = o;
    }
}

// ------- f32 [K][N] -> bf16 [Npad][K] transpose+convert (zero padded) -------
__global__ __launch_bounds__(256)
void cvt_transpose_kernel(const float* __restrict__ src,      // [K][N] f32
                          unsigned short* __restrict__ dst,   // [Npad][K] bf16
                          int K, int N, int Npad) {
    __shared__ float tile[32][33];
    const int kb = blockIdx.y * 32;
    const int nb = blockIdx.x * 32;
    const int tx = threadIdx.x & 31;
    const int ty = threadIdx.x >> 5;      // 0..7
#pragma unroll
    for (int i = ty; i < 32; i += 8) {
        int k = kb + i, n = nb + tx;
        tile[i][tx] = (k < K && n < N) ? src[(size_t)k * N + n] : 0.0f;
    }
    __syncthreads();
#pragma unroll
    for (int i = ty; i < 32; i += 8) {
        int n = nb + i, k = kb + tx;
        if (n < Npad && k < K)
            dst[(size_t)n * K + k] = f2bf(tile[tx][i]);
    }
}

// ---------------- tiled bf16 WMMA GEMM (async double-buffered) ----------------
// C[M,Ncols] = act(A[M,K] @ Bt^T + bias), Bt is [Npad>=BN*gridx][K] bf16.
// Block tile 128x256, 8 waves each computing a 64x64 tile (16 WMMAs / K-step).
// EPI==0: ReLU, store bf16. EPI==1: mask multiply, store f32 (gn<Ncols guarded).
union FragBF { v16bf v; uint4 q[2]; };

template<int EPI>
__global__ __launch_bounds__(256)
void gemm_bf16_wmma(const unsigned short* __restrict__ A,    // [M][K]
                    const unsigned short* __restrict__ Bt,   // [Npad][K]
                    const float* __restrict__ bias,
                    unsigned short* __restrict__ Cb,
                    float* __restrict__ Cf,
                    const unsigned char* __restrict__ mask,
                    int M, int Ncols, int K) {
    __shared__ __align__(16) unsigned short lsA[2][BM * LDSP];  // [m][k]
    __shared__ __align__(16) unsigned short lsB[2][BN * LDSP];  // [n][k]

    const int t    = threadIdx.x;
    const int lane = t & 31;
    const int wave = t >> 5;
    const int wm   = wave >> 2;      // 0..1  -> 64 rows each
    const int wn   = wave & 3;       // 0..3  -> 64 cols each
    const int hi   = lane >> 4;      // 0/1
    const int l16  = lane & 15;

    const int m_blk = blockIdx.y * BM;
    const int n_blk = blockIdx.x * BN;

    // A loader: 2 threads per row   (128 rows x 32 cols, 32B per thread)
    const int ar = t >> 1;
    const int ac = (t & 1) * 16;
    // B loader: 1 thread per row    (256 rows x 32 cols, 64B per thread)
    const unsigned offA = (unsigned)(((size_t)(m_blk + ar) * K + ac) * 2);
    const unsigned offB = (unsigned)(((size_t)(n_blk + t) * K) * 2);
    const unsigned laA[2] = { lds_off(&lsA[0][ar * LDSP + ac]),
                              lds_off(&lsA[1][ar * LDSP + ac]) };
    const unsigned laB[2] = { lds_off(&lsB[0][t * LDSP]),
                              lds_off(&lsB[1][t * LDSP]) };

    auto stage = [&](int buf, int k0) {
        const unsigned kb2 = (unsigned)k0 * 2u;
        async_copy_b128<0 >(laA[buf], offA + kb2, A);
        async_copy_b128<16>(laA[buf], offA + kb2, A);
        async_copy_b128<0 >(laB[buf], offB + kb2, Bt);
        async_copy_b128<16>(laB[buf], offB + kb2, Bt);
        async_copy_b128<32>(laB[buf], offB + kb2, Bt);
        async_copy_b128<48>(laB[buf], offB + kb2, Bt);
    };

    v8f zero = {};
    v8f acc[4][4];
#pragma unroll
    for (int i = 0; i < 4; ++i)
#pragma unroll
        for (int j = 0; j < 4; ++j) acc[i][j] = zero;

    const int nK = K / BK;

    // prologue: stage tile 0 into buffer 0
    stage(0, 0);
    wait_async();
    __syncthreads();

    for (int kt = 0; kt < nK; ++kt) {
        const int cur = kt & 1;
        if (kt + 1 < nK) stage(cur ^ 1, (kt + 1) * BK);   // overlap DMA w/ WMMA

        // ---- fragments + 16 WMMAs from buffer `cur` ----
        const int ka = hi * 8;    // lane<16: K 0..7/16..23 ; lane>=16: K 8..15/24..31
        const int kb = hi * 8;    // lane<16: K 0..7/16..23 ; lane>=16: K 8..15/24..31
        FragBF b[4];
#pragma unroll
        for (int j = 0; j < 4; ++j) {
            const unsigned short* base = &lsB[cur][(wn * 64 + j * 16 + l16) * LDSP + kb];
            b[j].q[0] = *reinterpret_cast<const uint4*>(base);
            b[j].q[1] = *reinterpret_cast<const uint4*>(base + 16);
        }
#pragma unroll
        for (int i = 0; i < 4; ++i) {
            FragBF a;
            const unsigned short* base = &lsA[cur][(wm * 64 + i * 16 + l16) * LDSP];
            a.q[0] = *reinterpret_cast<const uint4*>(base + ka);
            a.q[1] = *reinterpret_cast<const uint4*>(base + ka + 16);
#pragma unroll
            for (int j = 0; j < 4; ++j)
                acc[i][j] = __builtin_amdgcn_wmma_f32_16x16x32_bf16(
                    false, a.v, false, b[j].v, (short)0, acc[i][j], false, false);
        }

        // own async copies done + all waves synced -> next buffer valid, and
        // everyone finished reading `cur` before it is restaged.
        wait_async();
        __syncthreads();
    }

    // ---- epilogue ----
#pragma unroll
    for (int i = 0; i < 4; ++i) {
        const int gm_base = m_blk + wm * 64 + i * 16 + hi * 8;
#pragma unroll
        for (int j = 0; j < 4; ++j) {
            const int gn = n_blk + wn * 64 + j * 16 + l16;
            if (EPI == 0) {
                const float bv = bias[gn];
#pragma unroll
                for (int r = 0; r < 8; ++r) {
                    float v = acc[i][j][r] + bv;
                    v = v > 0.f ? v : 0.f;
                    Cb[(size_t)(gm_base + r) * Ncols + gn] = f2bf(v);
                }
            } else {
                if (gn < Ncols) {
                    const float bv = bias[gn];
#pragma unroll
                    for (int r = 0; r < 8; ++r) {
                        const int gm = gm_base + r;
                        float v = acc[i][j][r] + bv;
                        v *= (mask[gm] != 0) ? 1.0f : 0.0f;
                        Cf[(size_t)gm * Ncols + gn] = v;
                    }
                }
            }
        }
    }
}

// ---------------- host launcher ----------------
extern "C" void kernel_launch(void* const* d_in, const int* in_sizes, int n_in,
                              void* d_out, int out_size, void* d_ws, size_t ws_size,
                              hipStream_t stream) {
    (void)in_sizes; (void)n_in; (void)out_size; (void)ws_size;

    const float*         features   = (const float*)d_in[0];
    const unsigned char* valid_mask = (const unsigned char*)d_in[1];  // bool per ROI
    const float* W1 = (const float*)d_in[2];
    const float* b1 = (const float*)d_in[3];
    const float* W2 = (const float*)d_in[4];
    const float* b2 = (const float*)d_in[5];
    const float* W3 = (const float*)d_in[6];
    const float* b3 = (const float*)d_in[7];

    const int M    = 16 * 512;       // 8192 rows
    const int K1   = 7 * 7 * 256;    // 12544
    const int HID  = 1024;
    const int NC   = 81;
    const int NC_P = 256;            // padded class dim (one BN-wide tile)

    // workspace carve-out (256B aligned)
    char* ws = (char*)d_ws;
    size_t off = 0;
    auto carve = [&](size_t bytes) -> char* {
        char* p = ws + off;
        off = (off + bytes + 255) & ~(size_t)255;
        return p;
    };
    unsigned short* Xb  = (unsigned short*)carve((size_t)M * K1 * 2);      // [M][K1]
    unsigned short* W1t = (unsigned short*)carve((size_t)HID * K1 * 2);    // [1024][12544]
    unsigned short* W2t = (unsigned short*)carve((size_t)HID * HID * 2);   // [1024][1024]
    unsigned short* W3t = (unsigned short*)carve((size_t)NC_P * HID * 2);  // [256][1024]
    unsigned short* H1b = (unsigned short*)carve((size_t)M * HID * 2);     // [M][1024]
    unsigned short* H2b = (unsigned short*)carve((size_t)M * HID * 2);     // [M][1024]

    // features: straight convert
    {
        long long n4 = ((long long)M * K1) / 4;
        int blocks = (int)((n4 + 255) / 256);
        cvt_f32_bf16_kernel<<<blocks, 256, 0, stream>>>(features, Xb, n4);
    }
    // weights: convert + transpose to [N][K] (zero-padded rows for layer 3)
    cvt_transpose_kernel<<<dim3(HID / 32, K1 / 32),   256, 0, stream>>>(W1, W1t, K1,  HID, HID);
    cvt_transpose_kernel<<<dim3(HID / 32, HID / 32),  256, 0, stream>>>(W2, W2t, HID, HID, HID);
    cvt_transpose_kernel<<<dim3(NC_P / 32, HID / 32), 256, 0, stream>>>(W3, W3t, HID, NC,  NC_P);

    dim3 blk(256);
    // L1: relu(X @ W1 + b1) -> H1 (bf16)
    gemm_bf16_wmma<0><<<dim3(HID / BN, M / BM), blk, 0, stream>>>(
        Xb, W1t, b1, H1b, nullptr, nullptr, M, HID, K1);
    // L2: relu(H1 @ W2 + b2) -> H2 (bf16)
    gemm_bf16_wmma<0><<<dim3(HID / BN, M / BM), blk, 0, stream>>>(
        H1b, W2t, b2, H2b, nullptr, nullptr, M, HID, HID);
    // L3: (H2 @ W3 + b3) * mask -> out (f32, N=81 guarded in epilogue)
    gemm_bf16_wmma<1><<<dim3(1, M / BM), blk, 0, stream>>>(
        H2b, W3t, b3, nullptr, (float*)d_out, valid_mask, M, NC, HID);
}
